// GraphDecoder_63393717289218
// MI455X (gfx1250) — matrix operand order repaired and checked by reading, not compiled, not run on hardware
//
#include <hip/hip_runtime.h>
#include <hip/hip_bf16.h>

// ---------------------------------------------------------------------------
// Problem constants (reference: L=8, S=128, D=512, H=8, HD=64, NL=3)
// ---------------------------------------------------------------------------
#define LL 8
#define SS 128
#define DD 512
#define HH 8
#define HDIM 64
#define NLAYER 3
#define NTOK (LL * SS)       // 1024 tokens
#define EPSLN 1e-5f

typedef __bf16 v16bf __attribute__((ext_vector_type(16)));
typedef float  v8f   __attribute__((ext_vector_type(8)));
typedef float  v4f   __attribute__((ext_vector_type(4)));

// ---------------------------------------------------------------------------
// Generic WMMA GEMM:  C[M][N] = act( A[M][K] @ W[N][K]^T + bias[N] )
// Block = 256 threads = 8 waves. Block tile 128(rows) x 64(cols).
// Register-prefetch + double-buffered LDS pipeline; bf16 operands, f32 acc.
// All 4 B-fragments preloaded into live registers before the WMMA chain so
// LDS waits are graduated instead of dscnt==0 per tile.
// M % 128 == 0, N % 64 == 0, K % 32 == 0 at all call sites.
// ---------------------------------------------------------------------------
__global__ __launch_bounds__(256) void gemm_wmma(
    const float* __restrict__ A, const float* __restrict__ W,
    const float* __restrict__ bias, float* __restrict__ C,
    int M, int N, int K, int act)
{
  __shared__ __bf16 Asm[2][128][36];   // [buf][row][K32], stride 36 (18-bank skew)
  __shared__ __bf16 Wsm[2][64][36];    // [buf][col][K32]

  const int tid   = threadIdx.x;
  const int wave  = tid >> 5;
  const int lane  = tid & 31;
  const int lrow  = lane & 15;
  const int khalf = lane >> 4;
  const int rowBase = blockIdx.y * 128;
  const int colBase = blockIdx.x * 64;

  // staging assignment: A-tile 128x32 -> thread: row tid>>1, 16 floats (half tid&1)
  //                     W-tile  64x32 -> thread: row tid&63, 8 floats (chunk tid>>6)
  const int sr = tid >> 1, sh = tid & 1;
  const int wr = tid & 63, wc = (tid >> 6) * 8;

  const float* aRow = A + (size_t)(rowBase + sr) * K + sh * 16;
  const float* wRow = W + (size_t)(colBase + wr) * K + wc;

  v4f a0, a1, a2, a3, w0, w1;                      // register prefetch buffers
  auto loadStage = [&](int k0) {                   // issue all global loads
    const v4f* ap = (const v4f*)(aRow + k0);
    a0 = ap[0]; a1 = ap[1]; a2 = ap[2]; a3 = ap[3];
    const v4f* wp = (const v4f*)(wRow + k0);
    w0 = wp[0]; w1 = wp[1];
  };
  auto storeStage = [&](int buf) {                 // cvt f32->bf16, write LDS
    #pragma unroll
    for (int e = 0; e < 4; ++e) {
      Asm[buf][sr][sh * 16 + e]      = (__bf16)a0[e];
      Asm[buf][sr][sh * 16 + 4 + e]  = (__bf16)a1[e];
      Asm[buf][sr][sh * 16 + 8 + e]  = (__bf16)a2[e];
      Asm[buf][sr][sh * 16 + 12 + e] = (__bf16)a3[e];
      Wsm[buf][wr][wc + e]           = (__bf16)w0[e];
      Wsm[buf][wr][wc + 4 + e]       = (__bf16)w1[e];
    }
  };

  v8f c[4];
  #pragma unroll
  for (int t = 0; t < 4; ++t) c[t] = (v8f){0.f,0.f,0.f,0.f,0.f,0.f,0.f,0.f};

  loadStage(0);
  storeStage(0);
  const int steps = K >> 5;
  int buf = 0;
  for (int s = 0; s < steps; ++s) {
    __syncthreads();
    if (s + 1 < steps) loadStage((s + 1) << 5);    // global loads overlap WMMAs

    // ----- A fragment (16x32) from LDS, contiguous k = khalf*16 + e -----
    v16bf af;
    {
      const __bf16* ap = &Asm[buf][wave * 16 + lrow][0];
      #pragma unroll
      for (int e = 0; e < 16; ++e) {
        af[e] = ap[khalf * 16 + e];
      }
    }
    // ----- preload ALL 4 B fragments (forced-live), then WMMA chain -----
    v16bf b0v, b1v, b2v, b3v;
    {
      const __bf16* p0 = &Wsm[buf][ 0 + lrow][khalf * 16];
      const __bf16* p1 = &Wsm[buf][16 + lrow][khalf * 16];
      const __bf16* p2 = &Wsm[buf][32 + lrow][khalf * 16];
      const __bf16* p3 = &Wsm[buf][48 + lrow][khalf * 16];
      #pragma unroll
      for (int e = 0; e < 16; ++e) {
        b0v[e] = p0[e]; b1v[e] = p1[e]; b2v[e] = p2[e]; b3v[e] = p3[e];
      }
    }
    c[0] = __builtin_amdgcn_wmma_f32_16x16x32_bf16(false, af, false, b0v, (short)0, c[0], false, false);
    c[1] = __builtin_amdgcn_wmma_f32_16x16x32_bf16(false, af, false, b1v, (short)0, c[1], false, false);
    c[2] = __builtin_amdgcn_wmma_f32_16x16x32_bf16(false, af, false, b2v, (short)0, c[2], false, false);
    c[3] = __builtin_amdgcn_wmma_f32_16x16x32_bf16(false, af, false, b3v, (short)0, c[3], false, false);

    if (s + 1 < steps) storeStage(buf ^ 1);        // fill other buffer
    buf ^= 1;
  }

  // ----- epilogue: bias + optional relu, store f32 -----
  #pragma unroll
  for (int t = 0; t < 4; ++t) {
    const int col = colBase + t * 16 + lrow;
    const float bv = bias[col];
    #pragma unroll
    for (int r = 0; r < 8; ++r) {
      const int row = rowBase + wave * 16 + r + khalf * 8;
      float v = c[t][r] + bv;
      if (act) v = fmaxf(v, 0.f);
      C[(size_t)row * N + col] = v;
    }
  }
}

// ---------------------------------------------------------------------------
// Attention over axis L (=8) for each (s, h). qkv: [NTOK][3*DD] with token
// t = l*SS + s; q at col h*64, k at 512 + h*64, v at 1024 + h*64.
// One thread per (s,h): 8x8 scores -> softmax -> 8x64 output. float4 walks.
// ---------------------------------------------------------------------------
__global__ __launch_bounds__(256) void attn_kernel(
    const float* __restrict__ qkv, float* __restrict__ out)
{
  const int idx = blockIdx.x * 256 + threadIdx.x;   // 0..1023
  const int s = idx >> 3;
  const int h = idx & 7;
  const int TD = 3 * DD;

  float sc[LL][LL];
  #pragma unroll
  for (int l = 0; l < LL; ++l) {
    const v4f* q = (const v4f*)(qkv + (size_t)(l * SS + s) * TD + h * HDIM);
    #pragma unroll
    for (int m = 0; m < LL; ++m) {
      const v4f* k = (const v4f*)(qkv + (size_t)(m * SS + s) * TD + DD + h * HDIM);
      float acc = 0.f;
      #pragma unroll
      for (int d = 0; d < HDIM / 4; ++d) {
        v4f qv = q[d], kv = k[d];
        acc += qv[0]*kv[0] + qv[1]*kv[1] + qv[2]*kv[2] + qv[3]*kv[3];
      }
      sc[l][m] = acc * 0.125f;                      // 1/sqrt(64)
    }
  }
  #pragma unroll
  for (int l = 0; l < LL; ++l) {
    float mx = sc[l][0];
    #pragma unroll
    for (int m = 1; m < LL; ++m) mx = fmaxf(mx, sc[l][m]);
    float sum = 0.f;
    #pragma unroll
    for (int m = 0; m < LL; ++m) { sc[l][m] = __expf(sc[l][m] - mx); sum += sc[l][m]; }
    const float inv = 1.f / sum;
    #pragma unroll
    for (int m = 0; m < LL; ++m) sc[l][m] *= inv;
  }
  #pragma unroll
  for (int l = 0; l < LL; ++l) {
    for (int d4 = 0; d4 < HDIM / 4; ++d4) {
      v4f acc = (v4f){0.f, 0.f, 0.f, 0.f};
      #pragma unroll
      for (int m = 0; m < LL; ++m) {
        const v4f vv = ((const v4f*)(qkv + (size_t)(m * SS + s) * TD + 2 * DD + h * HDIM))[d4];
        acc += sc[l][m] * vv;
      }
      ((v4f*)(out + (size_t)(l * SS + s) * DD + h * HDIM))[d4] = acc;
    }
  }
}

// ---------------------------------------------------------------------------
// y = LayerNorm(x (+ res)) * w + b over last dim D=512. One block per token.
// ---------------------------------------------------------------------------
__global__ __launch_bounds__(256) void ln_kernel(
    const float* __restrict__ x, const float* __restrict__ res,
    const float* __restrict__ w, const float* __restrict__ b,
    float* __restrict__ y)
{
  const int t = blockIdx.x;
  const int tid = threadIdx.x;
  const float* xr = x + (size_t)t * DD;
  float v0 = xr[tid], v1 = xr[tid + 256];
  if (res) {
    const float* rr = res + (size_t)t * DD;
    v0 += rr[tid]; v1 += rr[tid + 256];
  }
  float s = v0 + v1, s2 = v0 * v0 + v1 * v1;
  #pragma unroll
  for (int o = 16; o > 0; o >>= 1) {
    s  += __shfl_xor(s,  o, 32);
    s2 += __shfl_xor(s2, o, 32);
  }
  __shared__ float rs[8], rq[8];
  const int wv = tid >> 5, ln = tid & 31;
  if (ln == 0) { rs[wv] = s; rq[wv] = s2; }
  __syncthreads();
  float ts = 0.f, tq = 0.f;
  #pragma unroll
  for (int i = 0; i < 8; ++i) { ts += rs[i]; tq += rq[i]; }
  const float mean = ts * (1.f / DD);
  const float var  = tq * (1.f / DD) - mean * mean;
  const float rstd = rsqrtf(var + EPSLN);
  y[(size_t)t * DD + tid]       = (v0 - mean) * rstd * w[tid]       + b[tid];
  y[(size_t)t * DD + tid + 256] = (v1 - mean) * rstd * w[tid + 256] + b[tid + 256];
}

// ---------------------------------------------------------------------------
// Relation einsum: out[b,i,j,k] = sum_d (sub[b,j,d]*vsub[b,i,d]) *
//                                       (obj[b,k,d]*vobj[b,i,d])
// One block per (b,i): 128x128 output, K=512. 8 waves; wave owns 16 rows x
// 128 cols (8 C-tiles). Register-prefetch + double-buffered LDS; scaling
// fused into the f32->bf16 conversion at LDS-store time. B-fragments
// preloaded in two groups of 4 (forced-live) around the WMMA chains.
// ---------------------------------------------------------------------------
__global__ __launch_bounds__(256) void relation_kernel(
    const float* __restrict__ sub, const float* __restrict__ obj,
    const float* __restrict__ vsub, const float* __restrict__ vobj,
    float* __restrict__ out)
{
  const int bI = blockIdx.y;   // 0..7
  const int i  = blockIdx.x;   // 0..127

  __shared__ __bf16 As[2][128][36];   // [buf][row j][K32]
  __shared__ __bf16 Bs[2][128][36];   // [buf][col k][K32]

  const int tid   = threadIdx.x;
  const int wave  = tid >> 5;
  const int lane  = tid & 31;
  const int lrow  = lane & 15;
  const int khalf = lane >> 4;

  const int lr   = tid >> 1;   // staging row/col 0..127
  const int half = tid & 1;    // which 16-wide K chunk

  const float* spB = sub  + ((size_t)bI * SS + lr) * DD + half * 16;
  const float* opB = obj  + ((size_t)bI * SS + lr) * DD + half * 16;
  const float* vsB = vsub + ((size_t)(bI * SS + i)) * DD + half * 16;
  const float* voB = vobj + ((size_t)(bI * SS + i)) * DD + half * 16;

  v4f s0, s1, s2, s3, o0, o1, o2, o3, x0, x1, x2, x3, y0, y1, y2, y3;
  auto loadStage = [&](int k0) {
    const v4f* sp = (const v4f*)(spB + k0);
    const v4f* op = (const v4f*)(opB + k0);
    const v4f* xp = (const v4f*)(vsB + k0);
    const v4f* yp = (const v4f*)(voB + k0);
    s0 = sp[0]; s1 = sp[1]; s2 = sp[2]; s3 = sp[3];
    o0 = op[0]; o1 = op[1]; o2 = op[2]; o3 = op[3];
    x0 = xp[0]; x1 = xp[1]; x2 = xp[2]; x3 = xp[3];
    y0 = yp[0]; y1 = yp[1]; y2 = yp[2]; y3 = yp[3];
  };
  auto storeStage = [&](int buf) {
    #pragma unroll
    for (int e = 0; e < 4; ++e) {
      As[buf][lr][half * 16 + e]      = (__bf16)(s0[e] * x0[e]);
      As[buf][lr][half * 16 + 4 + e]  = (__bf16)(s1[e] * x1[e]);
      As[buf][lr][half * 16 + 8 + e]  = (__bf16)(s2[e] * x2[e]);
      As[buf][lr][half * 16 + 12 + e] = (__bf16)(s3[e] * x3[e]);
      Bs[buf][lr][half * 16 + e]      = (__bf16)(o0[e] * y0[e]);
      Bs[buf][lr][half * 16 + 4 + e]  = (__bf16)(o1[e] * y1[e]);
      Bs[buf][lr][half * 16 + 8 + e]  = (__bf16)(o2[e] * y2[e]);
      Bs[buf][lr][half * 16 + 12 + e] = (__bf16)(o3[e] * y3[e]);
    }
  };

  v8f c[8];
  #pragma unroll
  for (int t = 0; t < 8; ++t) c[t] = (v8f){0.f,0.f,0.f,0.f,0.f,0.f,0.f,0.f};

  loadStage(0);
  storeStage(0);
  const int steps = DD >> 5;   // 16
  int buf = 0;
  for (int s = 0; s < steps; ++s) {
    __syncthreads();
    if (s + 1 < steps) loadStage((s + 1) << 5);

    v16bf a;
    {
      const __bf16* ap = &As[buf][wave * 16 + lrow][0];
      #pragma unroll
      for (int e = 0; e < 16; ++e) {
        a[e] = ap[khalf * 16 + e];
      }
    }
    // ---- group 0: tiles 0..3 preloaded (forced-live), then WMMAs ----
    {
      v16bf b0v, b1v, b2v, b3v;
      const __bf16* p0 = &Bs[buf][ 0 + lrow][khalf * 16];
      const __bf16* p1 = &Bs[buf][16 + lrow][khalf * 16];
      const __bf16* p2 = &Bs[buf][32 + lrow][khalf * 16];
      const __bf16* p3 = &Bs[buf][48 + lrow][khalf * 16];
      #pragma unroll
      for (int e = 0; e < 16; ++e) {
        b0v[e] = p0[e]; b1v[e] = p1[e]; b2v[e] = p2[e]; b3v[e] = p3[e];
      }
      c[0] = __builtin_amdgcn_wmma_f32_16x16x32_bf16(false, a, false, b0v, (short)0, c[0], false, false);
      c[1] = __builtin_amdgcn_wmma_f32_16x16x32_bf16(false, a, false, b1v, (short)0, c[1], false, false);
      c[2] = __builtin_amdgcn_wmma_f32_16x16x32_bf16(false, a, false, b2v, (short)0, c[2], false, false);
      c[3] = __builtin_amdgcn_wmma_f32_16x16x32_bf16(false, a, false, b3v, (short)0, c[3], false, false);
    }
    // ---- group 1: tiles 4..7 ----
    {
      v16bf b0v, b1v, b2v, b3v;
      const __bf16* p0 = &Bs[buf][ 64 + lrow][khalf * 16];
      const __bf16* p1 = &Bs[buf][ 80 + lrow][khalf * 16];
      const __bf16* p2 = &Bs[buf][ 96 + lrow][khalf * 16];
      const __bf16* p3 = &Bs[buf][112 + lrow][khalf * 16];
      #pragma unroll
      for (int e = 0; e < 16; ++e) {
        b0v[e] = p0[e]; b1v[e] = p1[e]; b2v[e] = p2[e]; b3v[e] = p3[e];
      }
      c[4] = __builtin_amdgcn_wmma_f32_16x16x32_bf16(false, a, false, b0v, (short)0, c[4], false, false);
      c[5] = __builtin_amdgcn_wmma_f32_16x16x32_bf16(false, a, false, b1v, (short)0, c[5], false, false);
      c[6] = __builtin_amdgcn_wmma_f32_16x16x32_bf16(false, a, false, b2v, (short)0, c[6], false, false);
      c[7] = __builtin_amdgcn_wmma_f32_16x16x32_bf16(false, a, false, b3v, (short)0, c[7], false, false);
    }
    if (s + 1 < steps) storeStage(buf ^ 1);
    buf ^= 1;
  }

  // 67 MB stream-out: non-temporal stores (do not pollute L2 retention)
  const size_t base = (((size_t)bI * SS + i) * SS) * SS;
  #pragma unroll
  for (int t = 0; t < 8; ++t) {
    const int col = t * 16 + lrow;
    #pragma unroll
    for (int r = 0; r < 8; ++r) {
      const int row = wave * 16 + r + khalf * 8;
      __builtin_nontemporal_store(c[t][r], &out[base + (size_t)row * SS + col]);
    }
  }
}

// ---------------------------------------------------------------------------
// Host-side orchestration
// ---------------------------------------------------------------------------
extern "C" void kernel_launch(void* const* d_in, const int* in_sizes, int n_in,
                              void* d_out, int out_size, void* d_ws, size_t ws_size,
                              hipStream_t stream) {
  const float* x          = (const float*)d_in[0];
  const float* attn_in_w  = (const float*)d_in[1];
  const float* attn_in_b  = (const float*)d_in[2];
  const float* attn_out_w = (const float*)d_in[3];
  const float* attn_out_b = (const float*)d_in[4];
  const float* ln1_w      = (const float*)d_in[5];
  const float* ln1_b      = (const float*)d_in[6];
  const float* ln2_w      = (const float*)d_in[7];
  const float* ln2_b      = (const float*)d_in[8];
  const float* ff1_w      = (const float*)d_in[9];
  const float* ff1_b      = (const float*)d_in[10];
  const float* ff2_w      = (const float*)d_in[11];
  const float* ff2_b      = (const float*)d_in[12];
  const float* fln_w      = (const float*)d_in[13];
  const float* fln_b      = (const float*)d_in[14];
  const float* mlp_w0     = (const float*)d_in[15];
  const float* mlp_b0     = (const float*)d_in[16];
  const float* mlp_w1     = (const float*)d_in[17];
  const float* mlp_b1     = (const float*)d_in[18];
  const float* mlp_w2     = (const float*)d_in[19];
  const float* mlp_b2     = (const float*)d_in[20];

  float* ws = (float*)d_ws;
  const size_t TOKD = (size_t)NTOK * DD;          // 524288 floats
  float* h      = ws;                              // [1024][512]
  float* hN     = ws + TOKD;                       // [1024][512]
  float* tmpA   = ws + 2 * TOKD;                   // [1024][2048] (qkv / ff1)
  float* tmpB   = ws + 6 * TOKD;                   // [1024][512]
  float* attn_o = ws + 7 * TOKD;                   // [1024][512]
  float* proj[4] = { ws + 8 * TOKD, ws + 9 * TOKD, ws + 10 * TOKD, ws + 11 * TOKD };
  float* g1     = ws + 12 * TOKD;
  float* g2     = ws + 13 * TOKD;

  auto gemm = [&](const float* A, const float* W, const float* B, float* C,
                  int N, int K, int act) {
    dim3 grid(N / 64, NTOK / 128);
    gemm_wmma<<<grid, 256, 0, stream>>>(A, W, B, C, NTOK, N, K, act);
  };

  // h = x
  hipMemcpyAsync(h, x, TOKD * sizeof(float), hipMemcpyDeviceToDevice, stream);

  // ------------------- 3 encoder layers -------------------
  for (int l = 0; l < NLAYER; ++l) {
    const float* iw  = attn_in_w  + (size_t)l * 3 * DD * DD;
    const float* ib  = attn_in_b  + (size_t)l * 3 * DD;
    const float* ow  = attn_out_w + (size_t)l * DD * DD;
    const float* ob  = attn_out_b + (size_t)l * DD;
    const float* f1w = ff1_w + (size_t)l * 4 * DD * DD;
    const float* f1b = ff1_b + (size_t)l * 4 * DD;
    const float* f2w = ff2_w + (size_t)l * DD * 4 * DD;
    const float* f2b = ff2_b + (size_t)l * DD;

    gemm(h, iw, ib, tmpA, 3 * DD, DD, 0);                      // qkv [1024][1536]
    attn_kernel<<<dim3(4), 256, 0, stream>>>(tmpA, attn_o);    // [1024][512]
    gemm(attn_o, ow, ob, tmpB, DD, DD, 0);                     // out proj
    ln_kernel<<<NTOK, 256, 0, stream>>>(h, tmpB, ln1_w + (size_t)l * DD,
                                        ln1_b + (size_t)l * DD, h);
    gemm(h, f1w, f1b, tmpA, 4 * DD, DD, 1);                    // ff1 + relu
    gemm(tmpA, f2w, f2b, tmpB, DD, 4 * DD, 0);                 // ff2
    ln_kernel<<<NTOK, 256, 0, stream>>>(h, tmpB, ln2_w + (size_t)l * DD,
                                        ln2_b + (size_t)l * DD, h);
  }
  // final encoder norm
  ln_kernel<<<NTOK, 256, 0, stream>>>(h, nullptr, fln_w, fln_b, hN);

  // ------------------- 4 projection MLPs -------------------
  for (int p = 0; p < 4; ++p) {
    const size_t wo = (size_t)p * DD * DD, bo = (size_t)p * DD;
    gemm(hN, mlp_w0 + wo, mlp_b0 + bo, g1, DD, DD, 1);
    gemm(g1, mlp_w1 + wo, mlp_b1 + bo, g2, DD, DD, 1);
    gemm(g2, mlp_w2 + wo, mlp_b2 + bo, proj[p], DD, DD, 0);
  }

  // ------------------- relation tensor (8,128,128,128) -------------------
  relation_kernel<<<dim3(SS, LL), 256, 0, stream>>>(
      proj[0], proj[1], proj[2], proj[3], (float*)d_out);
}
